// NeuralMLPF2_87969520156962
// MI455X (gfx1250) — matrix operand order repaired and checked by reading, not compiled, not run on hardware
//
#include <hip/hip_runtime.h>
#include <hip/hip_bf16.h>
#include <math.h>

// ---- problem constants ----
#define Nrows 8192
#define Lpos  256
#define Dd    128
#define Bb    64
#define KEEP  64
#define HID   1024
#define IN_DIM 8321            // D + KEEP*D + 1
#define KPAD   8352            // IN_DIM rounded up to multiple of 32 (261 chunks)

typedef __attribute__((ext_vector_type(16))) __bf16 v16bf;
typedef __attribute__((ext_vector_type(8)))  float  v8f;

static __device__ __forceinline__ unsigned pack2bf(__bf16 a, __bf16 b) {
    unsigned short ua = __builtin_bit_cast(unsigned short, a);
    unsigned short ub = __builtin_bit_cast(unsigned short, b);
    return (unsigned)ua | ((unsigned)ub << 16);
}

// ---------------------------------------------------------------------------
// Kernel 1: per-row selection of first KEEP masked positions + log1p(count)
// ---------------------------------------------------------------------------
__global__ __launch_bounds__(256) void select_kernel(const int* __restrict__ mask,
                                                     const int* __restrict__ count,
                                                     int* __restrict__ sel,
                                                     float* __restrict__ logc) {
    int n = blockIdx.x * blockDim.x + threadIdx.x;
    if (n >= Nrows) return;
    int cnt = 0;
    const int* mrow = mask + (size_t)n * Lpos;
    for (int pos = 0; pos < Lpos && cnt < KEEP; ++pos) {
        if (mrow[pos] != 0) { sel[n * KEEP + cnt] = pos; ++cnt; }
    }
    for (int j = cnt; j < KEEP; ++j) sel[n * KEEP + j] = -1;
    logc[n] = log1pf((float)count[n]);
}

// ---------------------------------------------------------------------------
// Kernel 2: transpose W1 to k-contiguous layout, split fp32 -> bf16 hi/lo
// w1hi/w1lo: [HID][KPAD] bf16, padded K region zeroed.
// ---------------------------------------------------------------------------
__global__ __launch_bounds__(256) void split_w1_kernel(const float* __restrict__ W1,
                                                       __bf16* __restrict__ w1hi,
                                                       __bf16* __restrict__ w1lo) {
    int kk = blockIdx.x * blockDim.x + threadIdx.x;
    int c  = blockIdx.y;
    if (kk >= KPAD) return;
    float f = (kk < IN_DIM) ? W1[(size_t)kk * HID + c] : 0.0f;
    __bf16 hi = (__bf16)f;
    __bf16 lo = (__bf16)(f - (float)hi);
    size_t o = (size_t)c * KPAD + kk;
    w1hi[o] = hi;
    w1lo[o] = lo;
}

// ---------------------------------------------------------------------------
// Kernel 3: fused gather + GEMM1(bf16x2 split WMMA, fp32 acc) + GELU + GEMM2
// Block: 256 threads (8 waves). Block computes 16 rows x all 1024 columns.
// Wave w owns columns [w*128, w*128+128) = 8 WMMA tiles of 16x16.
// A tile is converted to bf16 hi/lo ONCE per block and stored in LDS already
// in this code's contiguous-k A-fragment layout (k = 16*half + s); lanes reload it with ds_load_b128.
// Double-buffered so each K-chunk needs a single workgroup barrier.
// ---------------------------------------------------------------------------
__global__ __launch_bounds__(256) void mlp_kernel(const float* __restrict__ q,
                                                  const float* __restrict__ kbuf,
                                                  const int*   __restrict__ batch_idx,
                                                  const int*   __restrict__ sel,
                                                  const float* __restrict__ logc,
                                                  const __bf16* __restrict__ w1hi,
                                                  const __bf16* __restrict__ w1lo,
                                                  const float* __restrict__ b1,
                                                  const float* __restrict__ W2,
                                                  const float* __restrict__ b2,
                                                  float* __restrict__ out) {
    // A-fragment buffers: [parity][lane][dword], 32B per lane row
    __shared__ __align__(32) unsigned ahi_lds[2][32][8];
    __shared__ __align__(32) unsigned alo_lds[2][32][8];
    __shared__ int   sel_s[16 * KEEP];    // selected positions for the 16 rows
    __shared__ int   bidx_s[16];
    __shared__ float logc_s[16];
    __shared__ float red[8][16];          // cross-wave reduction for GEMM2

    const int tid   = threadIdx.x;
    const int lane  = tid & 31;
    const int wave  = tid >> 5;
    const int row0  = blockIdx.x * 16;
    const int m     = lane & 15;          // C-frag column mapping
    const int half  = lane >> 4;
    const int colbase = wave * 128;

    // ---- stage per-row metadata ----
    if (tid < 16) {
        bidx_s[tid] = batch_idx[row0 + tid];
        logc_s[tid] = logc[row0 + tid];
    }
    for (int i = tid; i < 16 * KEEP; i += 256)
        sel_s[i] = sel[(size_t)row0 * KEEP + i];
    __syncthreads();

    // each thread owns A-tile elements (rm, kk) and (rm, kk+1), kk even
    const int rm  = tid >> 4;             // row within tile (2 elems per thread)
    const int kk  = (tid & 15) << 1;      // even k within chunk
    // contiguous-k A layout (k = 16*half + s): dest lane + dest dword
    const int dl  = rm + ((kk >> 4) << 4);
    const int dv  = (kk & 15) >> 1;

    v8f acc[8] = {};   // 8 column tiles, fp32 accumulators

    for (int kc = 0; kc < KPAD; kc += 32) {
        const int pb = (kc >> 5) & 1;

        // ---- gather 2 fp32 elements, split to bf16 hi/lo, store packed ----
        float f[2];
        #pragma unroll
        for (int s = 0; s < 2; ++s) {
            int gk = kc + kk + s;
            float v;
            if (gk < Dd) {
                v = q[(size_t)(row0 + rm) * Dd + gk];
            } else if (gk < Dd + KEEP * Dd) {
                int r  = gk - Dd;
                int j  = r >> 7;          // which kept key
                int d  = r & 127;         // dim within key
                int ip = sel_s[rm * KEEP + j];
                v = (ip >= 0)
                        ? kbuf[((size_t)bidx_s[rm] * Lpos + ip) * Dd + d]
                        : 0.0f;
            } else if (gk == IN_DIM - 1) {
                v = logc_s[rm];
            } else {
                v = 0.0f;                 // pad region
            }
            f[s] = v;
        }
        __bf16 h0 = (__bf16)f[0], h1 = (__bf16)f[1];
        __bf16 l0 = (__bf16)(f[0] - (float)h0);
        __bf16 l1 = (__bf16)(f[1] - (float)h1);
        ahi_lds[pb][dl][dv] = pack2bf(h0, h1);
        alo_lds[pb][dl][dv] = pack2bf(l0, l1);

        __syncthreads();   // fill(pb) complete before any wave reads it

        // ---- reload A fragment (already in WMMA layout): 32B per lane ----
        v16bf ahi = *(const v16bf*)&ahi_lds[pb][lane][0];
        v16bf alo = *(const v16bf*)&alo_lds[pb][lane][0];

        // ---- 8 column tiles: load B hi/lo frags (k-contiguous), 3 WMMAs ----
        #pragma unroll
        for (int t = 0; t < 8; ++t) {
            int col = colbase + t * 16 + m;           // B layout: n = lane%16
            size_t off = (size_t)col * KPAD + kc + (half << 4); // kbase=half*16
            v16bf bh = *(const v16bf*)(w1hi + off);
            v16bf bl = *(const v16bf*)(w1lo + off);
            acc[t] = __builtin_amdgcn_wmma_f32_16x16x32_bf16(
                         false, ahi, false, bh, (short)0, acc[t], false, false);
            acc[t] = __builtin_amdgcn_wmma_f32_16x16x32_bf16(
                         false, ahi, false, bl, (short)0, acc[t], false, false);
            acc[t] = __builtin_amdgcn_wmma_f32_16x16x32_bf16(
                         false, alo, false, bh, (short)0, acc[t], false, false);
        }
        // no second barrier: next chunk writes the other parity buffer
    }

    // ---- epilogue: bias + exact-erf GELU + dot with W2, fused GEMM2 ----
    const float invsqrt2 = 0.70710678118654752f;
    float p[8] = {0, 0, 0, 0, 0, 0, 0, 0};   // partial sums per C-frag row r
    #pragma unroll
    for (int t = 0; t < 8; ++t) {
        int col  = colbase + t * 16 + m;       // C layout: N = lane%16
        float bb = b1[col];
        float w2 = W2[col];
        #pragma unroll
        for (int r = 0; r < 8; ++r) {          // C layout: M = r + 8*half
            float x = acc[t][r] + bb;
            float h = 0.5f * x * (1.0f + erff(x * invsqrt2));
            p[r] += h * w2;
        }
    }
    // reduce across the 16 lanes of each half (same row, different columns)
    #pragma unroll
    for (int r = 0; r < 8; ++r) {
        p[r] += __shfl_xor(p[r], 1, 16);
        p[r] += __shfl_xor(p[r], 2, 16);
        p[r] += __shfl_xor(p[r], 4, 16);
        p[r] += __shfl_xor(p[r], 8, 16);
    }
    if ((lane & 15) == 0) {
        #pragma unroll
        for (int r = 0; r < 8; ++r)
            red[wave][r + 8 * half] = p[r];
    }
    __syncthreads();
    if (tid < 16) {
        float s = 0.0f;
        #pragma unroll
        for (int w = 0; w < 8; ++w) s += red[w][tid];
        out[row0 + tid] = s + b2[0];
    }
}

// ---------------------------------------------------------------------------
extern "C" void kernel_launch(void* const* d_in, const int* in_sizes, int n_in,
                              void* d_out, int out_size, void* d_ws, size_t ws_size,
                              hipStream_t stream) {
    const float* q         = (const float*)d_in[0];
    const float* kbuf      = (const float*)d_in[1];
    const int*   batch_idx = (const int*)d_in[2];
    const int*   mask      = (const int*)d_in[3];
    const int*   count     = (const int*)d_in[4];
    const float* W1        = (const float*)d_in[5];
    const float* b1        = (const float*)d_in[6];
    const float* W2        = (const float*)d_in[7];
    const float* b2        = (const float*)d_in[8];
    float* out = (float*)d_out;

    // workspace layout (256B aligned)
    char* ws = (char*)d_ws;
    size_t off = 0;
    int*    sel  = (int*)(ws + off);   off += (size_t)Nrows * KEEP * sizeof(int);   // 2 MB
    off = (off + 255) & ~(size_t)255;
    float*  logc = (float*)(ws + off); off += (size_t)Nrows * sizeof(float);        // 32 KB
    off = (off + 255) & ~(size_t)255;
    __bf16* w1hi = (__bf16*)(ws + off); off += (size_t)HID * KPAD * sizeof(__bf16); // ~17 MB
    off = (off + 255) & ~(size_t)255;
    __bf16* w1lo = (__bf16*)(ws + off); off += (size_t)HID * KPAD * sizeof(__bf16); // ~17 MB
    (void)ws_size; (void)n_in; (void)in_sizes; (void)out_size;

    select_kernel<<<(Nrows + 255) / 256, 256, 0, stream>>>(mask, count, sel, logc);
    split_w1_kernel<<<dim3((KPAD + 255) / 256, HID), 256, 0, stream>>>(W1, w1hi, w1lo);
    mlp_kernel<<<Nrows / 16, 256, 0, stream>>>(q, kbuf, batch_idx, sel, logc,
                                               w1hi, w1lo, b1, W2, b2, out);
}
